// Informer_25692494365465
// MI455X (gfx1250) — matrix operand order repaired and checked by reading, not compiled, not run on hardware
//
#include <hip/hip_runtime.h>

typedef __attribute__((ext_vector_type(16))) _Float16 v16h;
typedef __attribute__((ext_vector_type(8)))  float    v8f;

#define BM 128
#define BN 64
#define BK 32
#define LDSTR 40   // LDS row stride in halves (32 + 8 pad)

// low 32 bits of a generic LDS pointer == raw LDS byte offset (ISA aperture rule)
static __device__ __forceinline__ unsigned int lds_off(const void* p)
{
    return (unsigned int)(unsigned long long)p;
}

// ---------------------------------------------------------------------------
// Generic WMMA GEMM: C[M,N] = A[M,K](f16) * W[N,K](f16)^T + bias[N], epilogue op
// op: 0=none 1=GELU 2=+resid 3=BN+ELU 4=ReLU 5=ReLU(x+resid) 6=ReLU(ReLU(x)+resid)
// Double-buffered LDS, global->LDS via async copy (ASYNCcnt), 1 barrier / k-step
// ---------------------------------------------------------------------------
__global__ __launch_bounds__(256)
void gemm_nt_f16(const _Float16* __restrict__ A, const _Float16* __restrict__ W,
                 const float* __restrict__ bias,
                 float* __restrict__ C32, _Float16* __restrict__ C16,
                 const float* __restrict__ resid,
                 const float* __restrict__ scale, const float* __restrict__ shift,
                 int M, int N, int K, int op)
{
    __shared__ _Float16 As[2][BM][LDSTR];
    __shared__ _Float16 Bs[2][BN][LDSTR];

    const int tid  = threadIdx.x;
    const int wave = tid >> 5;
    const int lane = tid & 31;
    const int wm   = wave & 3;     // 4 waves along M
    const int wn   = wave >> 2;    // 2 waves along N
    const int g    = lane >> 4;
    const int r    = lane & 15;
    const int m0   = blockIdx.x * BM;
    const int n0   = blockIdx.y * BN;

    // per-lane async-copy assignment (uniform 3 x b128 per lane)
    const int arow = tid >> 1, aseg = tid & 1;   // A tile: 128 rows x 32 halves
    const int brow = tid >> 2, bseg = tid & 3;   // B tile:  64 rows x 32 halves
    const _Float16* aSrc = A + (size_t)(m0 + arow) * K + aseg * 16;
    const _Float16* bSrc = W + (size_t)(n0 + brow) * K + bseg * 8;

    v8f acc[2][2] = {};
    const int nk = K / BK;

    // prologue: tile 0 into buffer 0
    {
        unsigned int ad = lds_off(&As[0][arow][aseg * 16]);
        unsigned int bd = lds_off(&Bs[0][brow][bseg * 8]);
        asm volatile("global_load_async_to_lds_b128 %0, %1, off" :: "v"(ad),       "v"(aSrc)     : "memory");
        asm volatile("global_load_async_to_lds_b128 %0, %1, off" :: "v"(ad + 16u), "v"(aSrc + 8) : "memory");
        asm volatile("global_load_async_to_lds_b128 %0, %1, off" :: "v"(bd),       "v"(bSrc)     : "memory");
    }

    for (int i = 0; i < nk; ++i) {
        asm volatile("s_wait_asynccnt 0x0" ::: "memory");
        __syncthreads();   // tile i visible to all; tile i-1 compute finished everywhere
        if (i + 1 < nk) {  // prefetch tile i+1 into the other buffer (overlaps WMMAs)
            int nb = (i + 1) & 1;
            int k1 = (i + 1) * BK;
            unsigned int ad = lds_off(&As[nb][arow][aseg * 16]);
            unsigned int bd = lds_off(&Bs[nb][brow][bseg * 8]);
            asm volatile("global_load_async_to_lds_b128 %0, %1, off" :: "v"(ad),       "v"(aSrc + k1)     : "memory");
            asm volatile("global_load_async_to_lds_b128 %0, %1, off" :: "v"(ad + 16u), "v"(aSrc + k1 + 8) : "memory");
            asm volatile("global_load_async_to_lds_b128 %0, %1, off" :: "v"(bd),       "v"(bSrc + k1)     : "memory");
        }

        const int cur = i & 1;
        v16h af[2], bf[2];
#pragma unroll
        for (int tm = 0; tm < 2; ++tm) {
            const _Float16* ap = &As[cur][wm * 32 + tm * 16 + r][0];
#pragma unroll
            for (int j = 0; j < 8; ++j) af[tm][j]     = ap[16 * g + j];      // K = h+16g
#pragma unroll
            for (int j = 0; j < 8; ++j) af[tm][8 + j] = ap[16 * g + 8 + j];  // K = h+8+16g
        }
#pragma unroll
        for (int tn = 0; tn < 2; ++tn) {
            const _Float16* bp = &Bs[cur][wn * 32 + tn * 16 + r][16 * g];    // K = h+16g
#pragma unroll
            for (int j = 0; j < 16; ++j) bf[tn][j] = bp[j];
        }
#pragma unroll
        for (int tm = 0; tm < 2; ++tm)
#pragma unroll
            for (int tn = 0; tn < 2; ++tn)
                acc[tm][tn] = __builtin_amdgcn_wmma_f32_16x16x32_f16(
                    false, af[tm], false, bf[tn], (short)0, acc[tm][tn], false, false);
    }

#pragma unroll
    for (int tm = 0; tm < 2; ++tm)
#pragma unroll
        for (int tn = 0; tn < 2; ++tn) {
            int col = n0 + wn * 32 + tn * 16 + r;
            float bb = bias ? bias[col] : 0.0f;
#pragma unroll
            for (int i = 0; i < 8; ++i) {
                int row = m0 + wm * 32 + tm * 16 + i + 8 * g;
                size_t off = (size_t)row * N + col;
                float v = acc[tm][tn][i] + bb;
                if (op == 1)      v = 0.5f * v * (1.0f + erff(v * 0.70710678f));
                else if (op == 2) v += resid[off];
                else if (op == 3) { v = v * (scale[col] * 0.999995000037f) + shift[col];
                                    v = v > 0.0f ? v : (__expf(v) - 1.0f); }
                else if (op == 4) v = fmaxf(v, 0.0f);
                else if (op == 5) v = fmaxf(v + resid[off], 0.0f);
                else if (op == 6) { v = fmaxf(v, 0.0f); v = fmaxf(v + resid[off], 0.0f); }
                if (C32) C32[off] = v;
                if (C16) C16[off] = (_Float16)v;
            }
        }
}

// ---------------------------------------------------------------------------
__global__ void cvt_f16_kernel(const float* __restrict__ in, _Float16* __restrict__ out, long n)
{
    long i = (long)blockIdx.x * 256 + threadIdx.x;
    if (i < n) out[i] = (_Float16)in[i];
}

// token conv (circular, k=3, 7->512) + positional + temporal sinusoid embeddings
__global__ __launch_bounds__(256)
void embed_kernel(const float* __restrict__ xe, const float* __restrict__ tokw,
                  const float* __restrict__ tokb, const int* __restrict__ marks,
                  float* __restrict__ X32, _Float16* __restrict__ X16, int L)
{
    int bl = blockIdx.x;
    int b = bl / L, l = bl % L;
    __shared__ float xin[3][7];
    __shared__ int mk[4];
    if (threadIdx.x < 21) {
        int t = threadIdx.x / 7, e = threadIdx.x % 7;
        int lm = (l + t - 1 + L) % L;
        xin[t][e] = xe[((size_t)b * L + lm) * 7 + e];
    }
    if (threadIdx.x < 4) mk[threadIdx.x] = marks[((size_t)b * L + l) * 4 + threadIdx.x];
    __syncthreads();
    for (int dm = threadIdx.x; dm < 512; dm += 256) {
        float acc = tokb[dm];
        for (int t = 0; t < 3; ++t)
            for (int e = 0; e < 7; ++e)
                acc += xin[t][e] * tokw[(dm * 7 + e) * 3 + t];
        int p = dm >> 1;
        float freq = __expf(-(float)(2 * p) * (9.210340371976184f / 512.0f));
        float s = 0.0f;
        if (dm & 1) {
            s += __cosf(l * freq);
            for (int j = 0; j < 4; ++j) s += __cosf(mk[j] * freq);
        } else {
            s += __sinf(l * freq);
            for (int j = 0; j < 4; ++j) s += __sinf(mk[j] * freq);
        }
        acc += s;
        size_t off = ((size_t)b * L + l) * 512 + dm;
        X32[off] = acc; X16[off] = (_Float16)acc;
    }
}

// M[b,h,l] = max_u(q.k_samp) - sum_u(q.k_samp)/L   (one wave per (b,h,l))
__global__ __launch_bounds__(256)
void measure_kernel(const float* __restrict__ Q, const float* __restrict__ Kk,
                    const int* __restrict__ idx, float* __restrict__ Mout, int L, int U)
{
    int w = (blockIdx.x * 256 + threadIdx.x) >> 5;
    int lane = threadIdx.x & 31;
    if (w >= 16 * 8 * L) return;
    int l = w % L; int h = (w / L) & 7; int b = w / (L * 8);
    const float* q = Q + ((size_t)(b * L + l)) * 512 + h * 64;
    float q0 = q[lane], q1 = q[lane + 32];
    float mx = -1e30f, sm = 0.0f;
    for (int u = 0; u < U; ++u) {
        int ks = idx[l * U + u];
        const float* kr = Kk + ((size_t)(b * L + ks)) * 512 + h * 64;
        float p = q0 * kr[lane] + q1 * kr[lane + 32];
        for (int o = 16; o > 0; o >>= 1) p += __shfl_xor(p, o);
        mx = fmaxf(mx, p); sm += p;
    }
    if (lane == 0) Mout[(size_t)(b * 8 + h) * L + l] = mx - sm / (float)L;
}

// top-U indices of M per (b,h) — iterative argmax (one block per (b,h))
__global__ __launch_bounds__(256)
void topk_kernel(const float* __restrict__ Mv, int* __restrict__ mtop, int L, int U)
{
    __shared__ float vals[384];
    __shared__ float rv[256];
    __shared__ int   ri[256];
    int bh = blockIdx.x;
    const float* m = Mv + (size_t)bh * L;
    for (int i = threadIdx.x; i < L; i += 256) vals[i] = m[i];
    __syncthreads();
    for (int t = 0; t < U; ++t) {
        float bv = -1e30f; int bi = 0x7fffffff;
        for (int i = threadIdx.x; i < L; i += 256) {
            float v = vals[i];
            if (v > bv || (v == bv && i < bi)) { bv = v; bi = i; }
        }
        rv[threadIdx.x] = bv; ri[threadIdx.x] = bi;
        __syncthreads();
        for (int s = 128; s > 0; s >>= 1) {
            if (threadIdx.x < s) {
                if (rv[threadIdx.x + s] > rv[threadIdx.x] ||
                    (rv[threadIdx.x + s] == rv[threadIdx.x] && ri[threadIdx.x + s] < ri[threadIdx.x])) {
                    rv[threadIdx.x] = rv[threadIdx.x + s]; ri[threadIdx.x] = ri[threadIdx.x + s];
                }
            }
            __syncthreads();
        }
        if (threadIdx.x == 0) { mtop[bh * U + t] = ri[0]; vals[ri[0]] = -1e30f; }
        __syncthreads();
    }
}

__global__ void vmean_kernel(const float* __restrict__ V, float* __restrict__ vm, int L)
{
    int i = blockIdx.x * 256 + threadIdx.x;
    if (i >= 16 * 512) return;
    int b = i >> 9, d = i & 511;
    float s = 0.0f;
    for (int l = 0; l < L; ++l) s += V[((size_t)(b * L + l)) * 512 + d];
    vm[i] = s / (float)L;
}

__global__ void ctx_fill_kernel(const float* __restrict__ vm, _Float16* __restrict__ CTX, int L)
{
    size_t i = (size_t)blockIdx.x * 256 + threadIdx.x;
    size_t total = (size_t)16 * L * 512;
    if (i >= total) return;
    int d = (int)(i & 511);
    int b = (int)(i / ((size_t)L * 512));
    CTX[i] = (_Float16)vm[b * 512 + d];
}

// full attention for the selected queries; one wave per (b,h,u), two-pass softmax
__global__ __launch_bounds__(256)
void attn_top_kernel(const float* __restrict__ Q, const float* __restrict__ Kk,
                     const float* __restrict__ V, const int* __restrict__ mtop,
                     _Float16* __restrict__ CTX, int L, int U)
{
    __shared__ float sbuf[8][385];
    int wv = threadIdx.x >> 5, lane = threadIdx.x & 31;
    int gw = blockIdx.x * 8 + wv;
    if (gw >= 16 * 8 * U) return;
    int u = gw % U; int h = (gw / U) & 7; int b = gw / (U * 8);
    int lq = mtop[(b * 8 + h) * U + u];
    const float* q = Q + ((size_t)(b * L + lq)) * 512 + h * 64;
    float q0 = q[lane], q1 = q[lane + 32];
    float mx = -1e30f;
    for (int l = 0; l < L; ++l) {
        const float* kr = Kk + ((size_t)(b * L + l)) * 512 + h * 64;
        float p = q0 * kr[lane] + q1 * kr[lane + 32];
        for (int o = 16; o > 0; o >>= 1) p += __shfl_xor(p, o);
        p *= 0.125f;   // 1/sqrt(64)
        sbuf[wv][l] = p;
        mx = fmaxf(mx, p);
    }
    float sum = 0.0f, a0 = 0.0f, a1 = 0.0f;
    for (int l = 0; l < L; ++l) {
        float e = __expf(sbuf[wv][l] - mx);
        const float* vr = V + ((size_t)(b * L + l)) * 512 + h * 64;
        sum += e; a0 += e * vr[lane]; a1 += e * vr[lane + 32];
    }
    float inv = 1.0f / sum;
    _Float16* c = CTX + ((size_t)(b * L + lq)) * 512 + h * 64;
    c[lane] = (_Float16)(a0 * inv); c[lane + 32] = (_Float16)(a1 * inv);
}

__global__ __launch_bounds__(256)
void layernorm_kernel(const float* __restrict__ X, const float* __restrict__ gw,
                      const float* __restrict__ bw, float* __restrict__ O32,
                      _Float16* __restrict__ O16)
{
    __shared__ float s1[256], s2[256];
    int row = blockIdx.x, tid = threadIdx.x;
    const float* x = X + (size_t)row * 512;
    float v0 = x[tid], v1 = x[tid + 256];
    s1[tid] = v0 + v1; s2[tid] = v0 * v0 + v1 * v1;
    __syncthreads();
    for (int s = 128; s > 0; s >>= 1) {
        if (tid < s) { s1[tid] += s1[tid + s]; s2[tid] += s2[tid + s]; }
        __syncthreads();
    }
    float mean = s1[0] * (1.0f / 512.0f);
    float var  = s2[0] * (1.0f / 512.0f) - mean * mean;
    float inv  = rsqrtf(var + 1e-5f);
    float o0 = (v0 - mean) * inv * gw[tid] + bw[tid];
    float o1 = (v1 - mean) * inv * gw[tid + 256] + bw[tid + 256];
    size_t off = (size_t)row * 512;
    O32[off + tid] = o0; O32[off + tid + 256] = o1;
    if (O16) { O16[off + tid] = (_Float16)o0; O16[off + tid + 256] = (_Float16)o1; }
}

// im2col for conv-distill: A[(b*L+l)][c*3+t] = x[b,(l+t-1)%L,c]  (circular)
__global__ void gather_conv_kernel(const _Float16* __restrict__ X16, _Float16* __restrict__ A, int L)
{
    size_t i = (size_t)blockIdx.x * 256 + threadIdx.x;
    size_t total = (size_t)16 * L * 512 * 3;
    if (i >= total) return;
    int t = (int)(i % 3); size_t r = i / 3;
    int c = (int)(r % 512); size_t bl = r / 512;
    int l = (int)(bl % L); int b = (int)(bl / L);
    int lm = (l + t - 1 + L) % L;
    A[i] = X16[((size_t)(b * L + lm)) * 512 + c];
}

// maxpool k=3 stride=2 pad=-inf over y[B,L,512] -> x[B,L/2,512]
__global__ void pool_kernel(const float* __restrict__ Y, float* __restrict__ X32,
                            _Float16* __restrict__ X16, int L)
{
    int Lo = L / 2;
    size_t i = (size_t)blockIdx.x * 256 + threadIdx.x;
    size_t total = (size_t)16 * Lo * 512;
    if (i >= total) return;
    int c = (int)(i & 511); size_t bl = i >> 9;
    int lo = (int)(bl % Lo); int b = (int)(bl / Lo);
    float m = -1e30f;
    for (int d = -1; d <= 1; ++d) {
        int l = 2 * lo + d;
        if (l >= 0 && l < L) m = fmaxf(m, Y[((size_t)(b * L + l)) * 512 + c]);
    }
    size_t off = ((size_t)(b * Lo + lo)) * 512 + c;
    X32[off] = m; X16[off] = (_Float16)m;
}

// im2col for causal dilated conv k=2: A[(b*L+l)][c*2+t] = t? x[l] : x[l-dil] (0-pad)
__global__ void gather_tcn_kernel(const _Float16* __restrict__ X16, _Float16* __restrict__ A,
                                  int L, int dil)
{
    size_t i = (size_t)blockIdx.x * 256 + threadIdx.x;
    size_t total = (size_t)16 * L * 512 * 2;
    if (i >= total) return;
    int t = (int)(i & 1); size_t r = i >> 1;
    int c = (int)(r % 512); size_t bl = r / 512;
    int l = (int)(bl % L); int b = (int)(bl / L);
    int ls = t ? l : l - dil;
    A[i] = (ls >= 0) ? X16[((size_t)(b * L + ls)) * 512 + c] : (_Float16)0.0f;
}

// 4 projection heads on flat (128, 6144); one wave per output element
__global__ __launch_bounds__(256)
void heads_kernel(const float* __restrict__ X,
                  const float* __restrict__ h0w, const float* __restrict__ h0b,
                  const float* __restrict__ h1w, const float* __restrict__ h1b,
                  const float* __restrict__ h2w, const float* __restrict__ h2b,
                  const float* __restrict__ h3w, const float* __restrict__ h3b,
                  float* __restrict__ out)
{
    int gw = (blockIdx.x * 256 + threadIdx.x) >> 5;
    int lane = threadIdx.x & 31;
    if (gw >= 128 * 60) return;
    int col = gw % 60, row = gw / 60;
    const float* wgt; float bias;
    if (col < 6)       { wgt = h0w + (size_t)col * 6144;        bias = h0b[col]; }
    else if (col < 18) { int c = col - 6;  wgt = h1w + (size_t)c * 6144; bias = h1b[c]; }
    else if (col < 36) { int c = col - 18; wgt = h2w + (size_t)c * 6144; bias = h2b[c]; }
    else               { int c = col - 36; wgt = h3w + (size_t)c * 6144; bias = h3b[c]; }
    int b = row >> 3, l0 = (row & 7) * 12;
    float s = 0.0f;
    for (int k = lane; k < 6144; k += 32) {
        int lo = k >> 9, c = k & 511;
        s += X[((size_t)(b * 96 + l0 + lo)) * 512 + c] * wgt[k];
    }
    for (int o = 16; o > 0; o >>= 1) s += __shfl_xor(s, o);
    if (lane == 0) out[(size_t)row * 60 + col] = s + bias;
}

// ---------------------------------------------------------------------------
extern "C" void kernel_launch(void* const* d_in, const int* in_sizes, int n_in,
                              void* d_out, int out_size, void* d_ws, size_t ws_size,
                              hipStream_t stream)
{
    // inputs in setup_inputs() dict order
    const float* x_enc = (const float*)d_in[0];
    const float* tok_w = (const float*)d_in[2];
    const float* tok_b = (const float*)d_in[3];
    const float* wq = (const float*)d_in[4];  const float* bq = (const float*)d_in[5];
    const float* wk = (const float*)d_in[6];  const float* bk = (const float*)d_in[7];
    const float* wv = (const float*)d_in[8];  const float* bv = (const float*)d_in[9];
    const float* wo = (const float*)d_in[10]; const float* bo = (const float*)d_in[11];
    const float* c1w = (const float*)d_in[12]; const float* c1b = (const float*)d_in[13];
    const float* c2w = (const float*)d_in[14]; const float* c2b = (const float*)d_in[15];
    const float* n1g = (const float*)d_in[16]; const float* n1b = (const float*)d_in[17];
    const float* n2g = (const float*)d_in[18]; const float* n2b = (const float*)d_in[19];
    const float* cvw = (const float*)d_in[20]; const float* cvb = (const float*)d_in[21];
    const float* bng = (const float*)d_in[22]; const float* bnb = (const float*)d_in[23];
    const float* tw1 = (const float*)d_in[24]; const float* tb1 = (const float*)d_in[25];
    const float* tw2 = (const float*)d_in[26]; const float* tb2 = (const float*)d_in[27];
    const float* ng  = (const float*)d_in[28]; const float* nb  = (const float*)d_in[29];
    const float* h0w = (const float*)d_in[30]; const float* h0b = (const float*)d_in[31];
    const float* h1w = (const float*)d_in[32]; const float* h1b = (const float*)d_in[33];
    const float* h2w = (const float*)d_in[34]; const float* h2b = (const float*)d_in[35];
    const float* h3w = (const float*)d_in[36]; const float* h3b = (const float*)d_in[37];
    const int* marks = (const int*)d_in[38];
    const int* idxs[3] = { (const int*)d_in[40], (const int*)d_in[41], (const int*)d_in[42] };

    // workspace arena
    char* ws = (char*)d_ws;
    size_t off = 0;
    auto arena = [&](size_t bytes) -> char* {
        char* p = ws + off;
        off = (off + bytes + 255) & ~(size_t)255;
        return p;
    };
    const size_t MLMAX = (size_t)16 * 384;              // 6144 rows max
    float*    X32   = (float*)   arena(MLMAX * 512 * 4);
    _Float16* X16   = (_Float16*)arena(MLMAX * 512 * 2);
    float*    Q32   = (float*)   arena(MLMAX * 512 * 4);
    float*    K32   = (float*)   arena(MLMAX * 512 * 4);
    float*    V32   = (float*)   arena(MLMAX * 512 * 4);
    float*    A32   = (float*)   arena(MLMAX * 512 * 4);
    _Float16* CTX16 = (_Float16*)arena(MLMAX * 512 * 2);
    _Float16* FFN16 = (_Float16*)arena(MLMAX * 2048 * 2);
    _Float16* GAT16 = (_Float16*)arena(MLMAX * 1536 * 2);
    _Float16* W16   = (_Float16*)arena((size_t)2048 * 512 * 2);
    float*    MEAS  = (float*)   arena((size_t)16 * 8 * 384 * 4);
    int*      MTOP  = (int*)     arena((size_t)16 * 8 * 30 * 4);
    float*    VM    = (float*)   arena((size_t)16 * 512 * 4);
    (void)ws_size; (void)in_sizes; (void)n_in; (void)out_size;

    auto cvt = [&](const float* src, _Float16* dst, long n) {
        cvt_f16_kernel<<<(unsigned)((n + 255) / 256), 256, 0, stream>>>(src, dst, n);
    };
    auto gemm = [&](const _Float16* Aa, const _Float16* Ww, const float* bias,
                    float* C32p, _Float16* C16p, const float* resid,
                    const float* sc, const float* sh, int M, int N, int K, int op) {
        dim3 grid(M / BM, N / BN);
        gemm_nt_f16<<<grid, 256, 0, stream>>>(Aa, Ww, bias, C32p, C16p, resid, sc, sh, M, N, K, op);
    };

    // ---- embedding ----
    embed_kernel<<<16 * 384, 256, 0, stream>>>(x_enc, tok_w, tok_b, marks, X32, X16, 384);

    // ---- 3 encoder layers ----
    int L = 384;
    const int Us[3] = { 30, 30, 25 };
    for (int i = 0; i < 3; ++i) {
        int Mr = 16 * L;
        int U = Us[i];
        const size_t dd = (size_t)512 * 512;

        cvt(wq + i * dd, W16, 512 * 512);
        gemm(X16, W16, bq + i * 512, Q32, nullptr, nullptr, nullptr, nullptr, Mr, 512, 512, 0);
        cvt(wk + i * dd, W16, 512 * 512);
        gemm(X16, W16, bk + i * 512, K32, nullptr, nullptr, nullptr, nullptr, Mr, 512, 512, 0);
        cvt(wv + i * dd, W16, 512 * 512);
        gemm(X16, W16, bv + i * 512, V32, nullptr, nullptr, nullptr, nullptr, Mr, 512, 512, 0);

        // ProbSparse attention
        measure_kernel<<<16 * L, 256, 0, stream>>>(Q32, K32, idxs[i], MEAS, L, U);
        topk_kernel<<<16 * 8, 256, 0, stream>>>(MEAS, MTOP, L, U);
        vmean_kernel<<<(16 * 512 + 255) / 256, 256, 0, stream>>>(V32, VM, L);
        {
            size_t tot = (size_t)16 * L * 512;
            ctx_fill_kernel<<<(unsigned)((tot + 255) / 256), 256, 0, stream>>>(VM, CTX16, L);
        }
        attn_top_kernel<<<16 * U, 256, 0, stream>>>(Q32, K32, V32, MTOP, CTX16, L, U);

        // O-projection + residual, layernorm1
        cvt(wo + i * dd, W16, 512 * 512);
        gemm(CTX16, W16, bo + i * 512, A32, nullptr, X32, nullptr, nullptr, Mr, 512, 512, 2);
        layernorm_kernel<<<Mr, 256, 0, stream>>>(A32, n1g + i * 512, n1b + i * 512, X32, X16);

        // FFN (GELU) + residual, layernorm2
        cvt(c1w + (size_t)i * 2048 * 512, W16, (long)2048 * 512);
        gemm(X16, W16, c1b + i * 2048, nullptr, FFN16, nullptr, nullptr, nullptr, Mr, 2048, 512, 1);
        cvt(c2w + (size_t)i * 512 * 2048, W16, (long)512 * 2048);
        gemm(FFN16, W16, c2b + i * 512, A32, nullptr, X32, nullptr, nullptr, Mr, 512, 2048, 2);
        layernorm_kernel<<<Mr, 256, 0, stream>>>(A32, n2g + i * 512, n2b + i * 512, X32, X16);

        // conv distill (layers 0,1): im2col GEMM + BN + ELU, then maxpool s2
        if (i < 2) {
            size_t tot = (size_t)16 * L * 512 * 3;
            gather_conv_kernel<<<(unsigned)((tot + 255) / 256), 256, 0, stream>>>(X16, GAT16, L);
            cvt(cvw + (size_t)i * 512 * 1536, W16, (long)512 * 1536);
            gemm(GAT16, W16, cvb + i * 512, A32, nullptr, nullptr,
                 bng + i * 512, bnb + i * 512, Mr, 512, 1536, 3);
            size_t tot2 = (size_t)16 * (L / 2) * 512;
            pool_kernel<<<(unsigned)((tot2 + 255) / 256), 256, 0, stream>>>(A32, X32, X16, L);
            L /= 2;
        }
    }

    // ---- TCN blocks (L = 96) ----
    int Mr = 16 * L;                 // 1536
    float* xcur = X32;               // current x (f32), X16 mirrors it
    float* xalt = Q32;
    for (int t = 0; t < 3; ++t) {
        int dil = 1 << t;
        size_t tot = (size_t)16 * L * 512 * 2;
        gather_tcn_kernel<<<(unsigned)((tot + 255) / 256), 256, 0, stream>>>(X16, GAT16, L, dil);
        cvt(tw1 + (size_t)t * 512 * 1024, W16, (long)512 * 1024);
        gemm(GAT16, W16, tb1 + t * 512, nullptr, CTX16, nullptr, nullptr, nullptr, Mr, 512, 1024, 4);
        gather_tcn_kernel<<<(unsigned)((tot + 255) / 256), 256, 0, stream>>>(CTX16, GAT16, L, dil);
        cvt(tw2 + (size_t)t * 512 * 1024, W16, (long)512 * 1024);
        gemm(GAT16, W16, tb2 + t * 512, xalt, X16, xcur, nullptr, nullptr, Mr, 512, 1024, 6);
        float* tmp = xcur; xcur = xalt; xalt = tmp;
    }

    // ---- final layernorm + heads ----
    layernorm_kernel<<<Mr, 256, 0, stream>>>(xcur, ng, nb, xalt, nullptr);
    heads_kernel<<<(128 * 60 * 32 + 255) / 256, 256, 0, stream>>>(
        xalt, h0w, h0b, h1w, h1b, h2w, h2b, h3w, h3b, (float*)d_out);
}
